// BasicLSTM_12051678232884
// MI455X (gfx1250) — compile-verified, matrix operand order checked by reading, not run on hardware
//
#include <hip/hip_runtime.h>
#include <hip/hip_bf16.h>
#include <cstdint>

// ---------------------------------------------------------------------------
// Spiking LSTM (SpikingJelly LIF gates) for MI455X / gfx1250, wave32 + WMMA.
//
//   * One fused GEMM per step: [x_t | h] (32 x 2048, bf16) times W^T slice.
//   * 128 persistent workgroups; WG b owns hidden units [8b, 8b+8).
//     Its weight slice (4 gates x 8 units x 2048 K = 128KB bf16) is
//     pre-swizzled into WMMA B-fragment order by a prologue kernel and pulled
//     into LDS ONCE via the Tensor Data Mover, reused for all 1024 steps.
//   * Per step: stage x_t (f32->bf16) and h (bf16) into LDS A buffer,
//     4 waves x 64 K-steps of v_wmma_f32_16x16x32_bf16, LIF in registers,
//     cross-gate combine via a small LDS spike buffer, write h, grid barrier.
// ---------------------------------------------------------------------------

typedef __attribute__((ext_vector_type(16))) __bf16 v16bf;
typedef __attribute__((ext_vector_type(8)))  __bf16 v8bf;
typedef __attribute__((ext_vector_type(4)))  __bf16 v4bf;
typedef __attribute__((ext_vector_type(8)))  float  v8f;
typedef __attribute__((ext_vector_type(4)))  unsigned int u32x4;
typedef __attribute__((ext_vector_type(8)))  int          i32x8;
typedef __attribute__((ext_vector_type(4)))  int          i32x4;

#define T_STEPS 1024
#define NB      32          // batch
#define HID     1024        // hidden
#define KTOT    2048        // I + H
#define NWG     128         // workgroups (one per WGP; must all be resident)
#define HC      8           // hidden units per WG
#define BLK     128         // threads per block = 4 waves
#define NKK     64          // K-steps of 32 (KTOT/32)
#define A_STRIDE 2056       // bf16 elements per A row (2048 + 8 pad for banks)

#define WLDS_BYTES   (2 * NKK * 32 * 16 * 2)            // 131072: B-fragments
#define ASTAGE_BYTES (NB * A_STRIDE * 2)                // 131584
#define SPK_BYTES    (4 * NB * HC * 4)                  // 4096
#define SHMEM_BYTES  (WLDS_BYTES + ASTAGE_BYTES + SPK_BYTES)

#define V_TH 0.1f

// ---------------------------------------------------------------------------
// Prologue 1: swizzle Wx (4H x I, f32) and Wh (4H x H, f32) into bf16 WMMA
// B-fragment order:  wfrag[b][nt][kk][lane][e], 2^23 elements total.
// (lane,e) -> (n, k) per ISA 7.12.2 16-bit 32x16 B layout:
//   n = lane & 15;  khalf = (lane>=16)?8:0;  k = kk*32 + khalf + (e<8 ? e : e+8)
// column col = nt*16 + n maps to gate g = col>>3 (0:i 1:o 2:f 3:cs), unit u.
// ---------------------------------------------------------------------------
__global__ void prep_weights(const float* __restrict__ Wx,
                             const float* __restrict__ Wh,
                             __bf16* __restrict__ wfrag) {
    unsigned idx  = blockIdx.x * blockDim.x + threadIdx.x;   // < 1<<23
    unsigned e    = idx & 15u;
    unsigned lane = (idx >> 4) & 31u;
    unsigned kk   = (idx >> 9) & 63u;
    unsigned nt   = (idx >> 15) & 1u;
    unsigned b    = idx >> 16;
    unsigned khalf = (lane >= 16u) ? 8u : 0u;
    unsigned k    = kk * 32u + khalf + ((e < 8u) ? e : (e + 8u));
    unsigned col  = nt * 16u + (lane & 15u);
    unsigned g    = col >> 3;
    unsigned u    = col & 7u;
    unsigned row  = g * HID + b * HC + u;                    // row of [Wx;Wh]
    float v = (k < 1024u) ? Wx[row * 1024u + k]
                          : Wh[row * 1024u + (k - 1024u)];
    wfrag[idx] = (__bf16)v;
}

// Prologue 2: zero h buffer (32x1024 bf16 = 16384 dwords) + barrier counter.
__global__ void init_state(uint32_t* __restrict__ hbuf_u32,
                           unsigned* __restrict__ ctr) {
    int i = blockIdx.x * blockDim.x + threadIdx.x;   // 64*256 = 16384
    hbuf_u32[i] = 0u;
    if (i == 0) *ctr = 0u;
}

// ---------------------------------------------------------------------------
// Persistent recurrent kernel.
// ---------------------------------------------------------------------------
__global__ __launch_bounds__(BLK, 1)
void slstm_persistent(const float* __restrict__ x,     // [T, N, I] f32
                      const float* __restrict__ bx,    // [4H]
                      const float* __restrict__ bh,    // [4H]
                      const __bf16* __restrict__ wfrag,
                      __bf16* __restrict__ hbuf,       // [N, H] bf16
                      unsigned* __restrict__ bar,
                      float* __restrict__ out) {       // [T*N*H + 2*N*H] f32
    extern __shared__ __attribute__((aligned(16))) char smem_raw[];
    __bf16* Wlds   = (__bf16*)smem_raw;                          // B frags
    __bf16* Astage = (__bf16*)(smem_raw + WLDS_BYTES);           // [32][2056]
    float*  Sspk   = (float*)(smem_raw + WLDS_BYTES + ASTAGE_BYTES); // [4][32][8]

    const int tid  = threadIdx.x;
    const int b    = blockIdx.x;
    const int lane = tid & 31;
    const int w    = tid >> 5;          // wave 0..3
    const int mt   = w >> 1;            // batch tile (0..1)
    const int nt   = w & 1;             // column tile (0..1)
    const int ub   = b * HC;

    // ---- one-time: TDM-load this WG's pre-swizzled 128KB weight slice ----
    // D# per CDNA5 ISA ch.8: 2D tensor of 2-byte elements, 8192 x 8,
    // tile == tensor, contiguous (stride 8192), global -> LDS.
    if (w == 0) {
        const uint64_t ga = (uint64_t)(uintptr_t)(wfrag + (size_t)b * 65536);
        const uint32_t la = (uint32_t)(uintptr_t)Wlds;   // LDS addr[31:0]
        u32x4 g0 = {
            1u,                                              // count=1 (valid)
            la,                                              // lds_addr
            (unsigned)(ga & 0xFFFFFFFFu),                    // global_addr lo
            (unsigned)((ga >> 32) & 0x01FFFFFFu) | (2u << 30) // addr hi | type=2
        };
        i32x8 g1 = {
            (int)(1u << 16),          // workgroup_mask=0, data_size=1 (2B)
            (int)(8192u << 16),       // tensor_dim0[15:0]=8192  (bits 63:48)
            (int)(8u << 16),          // tensor_dim0 hi=0 | tensor_dim1 lo=8
            (int)(8192u << 16),       // tensor_dim1 hi=0 | tile_dim0=8192
            8,                        // tile_dim1=8, tile_dim2=0
            8192,                     // tensor_dim0_stride lo32
            0, 0                      // stride0 hi | stride1
        };
        i32x4 g2 = {0, 0, 0, 0};
        i32x4 g3 = {0, 0, 0, 0};
#if defined(__clang_major__) && (__clang_major__ >= 23)
        i32x8 g4 = {0, 0, 0, 0, 0, 0, 0, 0};
        __builtin_amdgcn_tensor_load_to_lds(g0, g1, g2, g3, g4, 0);
#else
        __builtin_amdgcn_tensor_load_to_lds(g0, g1, g2, g3, 0);
#endif
        __builtin_amdgcn_s_wait_tensorcnt((short)0);
    }

    // per-lane constants: which gate column this lane's D elements hold
    const int coln  = lane & 15;            // n within tile (also A-row % 16)
    const int col   = nt * 16 + coln;       // 0..31 within WG slice
    const int g     = col >> 3;             // 0:i 1:o 2:f 3:cs
    const int u     = col & 7;              // unit within slice
    const int grow  = g * HID + ub + u;
    const float bias = bx[grow] + bh[grow];
    const int khalf = (lane >= 16) ? 8 : 0;
    const int mhi   = 8 * (lane >> 4);      // D-layout M offset per lane half

    float vlif[8];                          // gate membrane, one per D VGPR
    #pragma unroll
    for (int r = 0; r < 8; ++r) vlif[r] = 0.f;

    float creg[2] = {0.f, 0.f};             // c,  owned elems p = 2*tid + j
    float vct[2]  = {0.f, 0.f};             // v_ct

    __syncthreads();                        // publish Wlds to all waves

    for (int t = 0; t < T_STEPS; ++t) {
        // ---- stage x_t (f32 -> bf16) into Astage[:, 0:1024] ----
        {
            const float* xs = x + (size_t)t * (NB * 1024);
            #pragma unroll 4
            for (int i = 0; i < 64; ++i) {
                int q = tid + i * BLK;           // vec4 index over 8192
                int m = q >> 8;                  // 256 vec4 per row
                int off = (q & 255) * 4;
                const float4 f = *(const float4*)(xs + m * 1024 + off);
                v4bf ov = { (__bf16)f.x, (__bf16)f.y, (__bf16)f.z, (__bf16)f.w };
                *(v4bf*)(Astage + m * A_STRIDE + off) = ov;
                if (t < T_STEPS - 1 && (q & 63) == 0)
                    __builtin_prefetch(xs + NB * 1024 + m * 1024 + off, 0, 1);
            }
        }
        // ---- stage h (bf16) into Astage[:, 1024:2048] ----
        {
            #pragma unroll 4
            for (int i = 0; i < 32; ++i) {
                int q = tid + i * BLK;           // vec8 index over 4096
                int m = q >> 7;                  // 128 vec8 per row
                int off = (q & 127) * 8;
                *(v8bf*)(Astage + m * A_STRIDE + 1024 + off) =
                    *(const v8bf*)(hbuf + m * HID + off);
            }
        }
        __syncthreads();

        // ---- GEMM: D(16x16) = A(16x32) x B(32x16) over 64 K-steps ----
        v8f acc = {0.f, 0.f, 0.f, 0.f, 0.f, 0.f, 0.f, 0.f};
        const __bf16* Wtile = Wlds + nt * (NKK * 512);
        const __bf16* Arow  = Astage + (mt * 16 + coln) * A_STRIDE + khalf;
        #pragma unroll 8
        for (int kk = 0; kk < NKK; ++kk) {
            const __bf16* ap = Arow + kk * 32;
            v8bf a0 = *(const v8bf*)(ap);        // K = kb+khalf   .. +7
            v8bf a1 = *(const v8bf*)(ap + 16);   // K = kb+khalf+16.. +23
            v16bf af = __builtin_shufflevector(a0, a1,
                0, 1, 2, 3, 4, 5, 6, 7, 8, 9, 10, 11, 12, 13, 14, 15);
            v16bf bfr = *(const v16bf*)(Wtile + kk * 512 + lane * 16);
            acc = __builtin_amdgcn_wmma_f32_16x16x32_bf16(
                false, af, false, bfr, (short)0, acc, false, false);
        }

        // ---- LIF on this wave's gates; publish spikes to LDS ----
        #pragma unroll
        for (int r = 0; r < 8; ++r) {
            int m = mt * 16 + r + mhi;                 // batch index
            float gv = acc[r] + bias;
            float v  = 0.5f * (vlif[r] + gv);          // charge (tau=2)
            float s  = (v >= V_TH) ? 1.f : 0.f;        // heaviside
            vlif[r]  = (1.f - s) * v;                  // hard reset
            Sspk[g * 256 + m * HC + u] = s;
        }
        __syncthreads();

        // ---- combine gates: c, v_ct LIF, h; each thread owns 2 (m,u) ----
        #pragma unroll
        for (int j = 0; j < 2; ++j) {
            int p = tid * 2 + j;                       // 0..255
            int m  = p >> 3;
            int uu = p & 7;
            float si  = Sspk[        p];
            float so  = Sspk[256  +  p];
            float sf  = Sspk[512  +  p];
            float scs = Sspk[768  +  p];
            float c = sf * creg[j] + si * scs;
            creg[j] = c;
            float v   = 0.5f * (vct[j] + c);
            float sct = (v >= V_TH) ? 1.f : 0.f;
            vct[j]    = (1.f - sct) * v;
            float h   = so * sct;
            out[(size_t)t * (NB * HID) + m * HID + ub + uu] = h;
            hbuf[m * HID + ub + uu] = (__bf16)h;
            if (t == T_STEPS - 1) {
                size_t tail = (size_t)T_STEPS * NB * HID;
                out[tail + m * HID + ub + uu]            = h;   // h_t
                out[tail + NB * HID + m * HID + ub + uu] = c;   // c_t
            }
        }

        // ---- grid-wide barrier (monotonic counter; all NWG resident) ----
        __threadfence();
        __syncthreads();
        if (tid == 0) {
            atomicAdd(bar, 1u);
            unsigned target = (unsigned)(t + 1) * NWG;
            while (__hip_atomic_load(bar, __ATOMIC_ACQUIRE,
                                     __HIP_MEMORY_SCOPE_AGENT) < target) {
                __builtin_amdgcn_s_sleep(1);
            }
        }
        __syncthreads();
    }
}

// ---------------------------------------------------------------------------
// Launch.  d_in order: inputs, Wx, bx, Wh, bh.  ws layout:
//   [0, 16MB)        : wfrag  (8,388,608 bf16, WMMA-fragment-ordered weights)
//   [16MB, +64KB)    : hbuf   (32x1024 bf16)
//   [+64KB, +4B)     : grid barrier counter
// ---------------------------------------------------------------------------
extern "C" void kernel_launch(void* const* d_in, const int* in_sizes, int n_in,
                              void* d_out, int out_size, void* d_ws, size_t ws_size,
                              hipStream_t stream) {
    const float* x  = (const float*)d_in[0];
    const float* Wx = (const float*)d_in[1];
    const float* bx = (const float*)d_in[2];
    const float* Wh = (const float*)d_in[3];
    const float* bh = (const float*)d_in[4];
    float* out = (float*)d_out;

    char* ws = (char*)d_ws;
    __bf16*   wfrag = (__bf16*)ws;
    __bf16*   hbuf  = (__bf16*)(ws + (size_t)16 * 1024 * 1024);
    unsigned* ctr   = (unsigned*)(ws + (size_t)16 * 1024 * 1024 + 65536);

    // 2^23 fragment elements, 256 threads each
    prep_weights<<<32768, 256, 0, stream>>>(Wx, Wh, wfrag);
    init_state<<<64, 256, 0, stream>>>((uint32_t*)hbuf, ctr);
    slstm_persistent<<<NWG, BLK, SHMEM_BYTES, stream>>>(
        x, bx, bh, wfrag, hbuf, ctr, out);
}
